// GCN_84378927497741
// MI455X (gfx1250) — matrix operand order repaired and checked by reading, not compiled, not run on hardware
//
#include <hip/hip_runtime.h>
#include <hip/hip_bf16.h>

// ---- CDNA5 WMMA types --------------------------------------------------
typedef __attribute__((ext_vector_type(16))) __bf16 v16bf;
typedef __attribute__((ext_vector_type(8)))  float  v8f;

#define IN_DIM 256
#define HID    64

// ---- Kernel 0: zero fill ----------------------------------------------
__global__ void gcn_zero_f32(float* __restrict__ p, long long n) {
    long long i = (long long)blockIdx.x * blockDim.x + threadIdx.x;
    long long stride = (long long)gridDim.x * blockDim.x;
    for (; i < n; i += stride) p[i] = 0.0f;
}

// ---- Kernel 1: pre-swizzle W1 into bf16 hi/lo wave-fragment order ------
// Fragment index t = ((colTile*8 + kstep)*32 + lane); each holds 16 bf16:
// element 2j / 2j+1  ->  B[K][n] with K = kstep*32 + 16*(lane/16) + 2j (+1),
// n = colTile*16 + (lane&15).
__global__ void gcn_prep_w1_frags(const float* __restrict__ W1,
                                  v16bf* __restrict__ bhi,
                                  v16bf* __restrict__ blo) {
    int t = blockIdx.x * blockDim.x + threadIdx.x;
    if (t >= 4 * 8 * 32) return;
    int lane = t & 31;
    int ks   = (t >> 5) & 7;
    int ct   = t >> 8;
    int half = lane >> 4;
    int n    = ct * 16 + (lane & 15);
    v16bf h, l;
#pragma unroll
    for (int j = 0; j < 8; ++j) {
        int K = ks * 32 + 16 * half + 2 * j;
        float v0 = W1[(size_t)K * HID + n];
        float v1 = W1[(size_t)(K + 1) * HID + n];
        __bf16 h0 = (__bf16)v0, h1 = (__bf16)v1;
        h[2 * j] = h0;
        h[2 * j + 1] = h1;
        l[2 * j] = (__bf16)(v0 - (float)h0);
        l[2 * j + 1] = (__bf16)(v1 - (float)h1);
    }
    bhi[t] = h;
    blo[t] = l;
}

// ---- Kernel 2: GEMM1  XW1 = X @ W1 + b1  via bf16 hi/lo split WMMA -----
// Block = 128 threads (4 waves). Wave w owns column tile w (16 cols);
// block covers 64 rows (4 row tiles of 16). K = 256 -> 8 k-steps of 32.
// acc += Ahi*Bhi + Alo*Bhi + Ahi*Blo  (fp32 accumulate).
__global__ __launch_bounds__(128) void gcn_gemm1_wmma(
    const float* __restrict__ X, const float* __restrict__ b1,
    const v16bf* __restrict__ bhi_frags, const v16bf* __restrict__ blo_frags,
    float* __restrict__ XW1, int N) {
    const int lane = threadIdx.x & 31;
    const int wave = threadIdx.x >> 5;   // column tile 0..3
    const int half = lane >> 4;          // 0/1
    const int mrow = lane & 15;
    const int rb   = blockIdx.x * 64;

    v8f zero = {0.f, 0.f, 0.f, 0.f, 0.f, 0.f, 0.f, 0.f};
    v8f acc[4];
#pragma unroll
    for (int rt = 0; rt < 4; ++rt) acc[rt] = zero;

    for (int ks = 0; ks < 8; ++ks) {
        const int k0 = ks * 32;
        const int fidx = (wave * 8 + ks) * 32 + lane;
        v16bf bhi = bhi_frags[fidx];     // 2x b128 load, L2-resident
        v16bf blo = blo_frags[fidx];
#pragma unroll
        for (int rt = 0; rt < 4; ++rt) {
            int row = rb + rt * 16 + mrow;
            row = row < N ? row : N - 1;            // clamp tail reads
            const float* xp = X + (size_t)row * IN_DIM + k0 + 16 * half;
            v16bf ahi, alo;
#pragma unroll
            for (int j = 0; j < 8; ++j) {
                int kp = 2 * j; // +16*half in xp
                float2 v = *(const float2*)(xp + kp);
                __bf16 h0 = (__bf16)v.x, h1 = (__bf16)v.y;
                ahi[2 * j] = h0;
                ahi[2 * j + 1] = h1;
                alo[2 * j] = (__bf16)(v.x - (float)h0);
                alo[2 * j + 1] = (__bf16)(v.y - (float)h1);
            }
            acc[rt] = __builtin_amdgcn_wmma_f32_16x16x32_bf16(
                false, ahi, false, bhi, (short)0, acc[rt], false, false);
            acc[rt] = __builtin_amdgcn_wmma_f32_16x16x32_bf16(
                false, alo, false, bhi, (short)0, acc[rt], false, false);
            acc[rt] = __builtin_amdgcn_wmma_f32_16x16x32_bf16(
                false, ahi, false, blo, (short)0, acc[rt], false, false);
        }
    }

    const int col = wave * 16 + (lane & 15);
    const float bias = b1[col];
#pragma unroll
    for (int rt = 0; rt < 4; ++rt) {
#pragma unroll
        for (int r = 0; r < 8; ++r) {
            int row = rb + rt * 16 + r + 8 * half;
            if (row < N) XW1[(size_t)row * HID + col] = acc[rt][r] + bias;
        }
    }
}

// ---- Kernel 3: SpMM1  Hraw[dst] += w * XW1[src]  (wave per edge) -------
__global__ void gcn_spmm1(const float* __restrict__ XW1,
                          const float* __restrict__ ew,
                          const int* __restrict__ esrc,
                          const int* __restrict__ edst,
                          float* __restrict__ H, int E) {
    int lane = threadIdx.x & 31;
    long long wid = ((long long)blockIdx.x * blockDim.x + threadIdx.x) >> 5;
    long long nw  = ((long long)gridDim.x * blockDim.x) >> 5;
    for (long long e = wid; e < E; e += nw) {
        int s = esrc[e], d = edst[e];
        float w = ew[e];
        const float* src = XW1 + (size_t)s * HID;
        float* dst = H + (size_t)d * HID;
        atomicAdd(&dst[lane],      w * src[lane]);
        atomicAdd(&dst[lane + 32], w * src[lane + 32]);
    }
}

// ---- Kernel 4: Z2 = relu(Hraw) @ W2 + b2  (wave per node) --------------
__global__ void gcn_relu_gemm2(const float* __restrict__ H,
                               const float* __restrict__ W2,
                               const float* __restrict__ b2,
                               float* __restrict__ Z2, int N) {
    int lane = threadIdx.x & 31;
    long long wid = ((long long)blockIdx.x * blockDim.x + threadIdx.x) >> 5;
    long long nw  = ((long long)gridDim.x * blockDim.x) >> 5;
    float w0 = W2[lane], w1 = W2[lane + 32];
    float bb = b2[0];
    for (long long i = wid; i < N; i += nw) {
        const float* h = H + (size_t)i * HID;
        float a = h[lane];      a = a > 0.f ? a : 0.f;
        float b = h[lane + 32]; b = b > 0.f ? b : 0.f;
        float acc = a * w0 + b * w1;
#pragma unroll
        for (int off = 16; off; off >>= 1) acc += __shfl_xor(acc, off, 32);
        if (lane == 0) Z2[i] = acc + bb;
    }
}

// ---- Kernel 5: SpMM2  out[dst] += w * Z2[src]  (thread per edge) -------
__global__ void gcn_spmm2(const float* __restrict__ Z2,
                          const float* __restrict__ ew,
                          const int* __restrict__ esrc,
                          const int* __restrict__ edst,
                          float* __restrict__ out, int E) {
    long long i = (long long)blockIdx.x * blockDim.x + threadIdx.x;
    long long stride = (long long)gridDim.x * blockDim.x;
    for (; i < E; i += stride)
        atomicAdd(&out[edst[i]], ew[i] * Z2[esrc[i]]);
}

// ---- Launcher ----------------------------------------------------------
extern "C" void kernel_launch(void* const* d_in, const int* in_sizes, int n_in,
                              void* d_out, int out_size, void* d_ws, size_t ws_size,
                              hipStream_t stream) {
    const float* X   = (const float*)d_in[0];
    const float* W1  = (const float*)d_in[1];
    const float* b1  = (const float*)d_in[2];
    const float* W2  = (const float*)d_in[3];
    const float* b2  = (const float*)d_in[4];
    const float* ew  = (const float*)d_in[5];
    const int*   esrc = (const int*)d_in[6];
    const int*   edst = (const int*)d_in[7];

    const int N = in_sizes[0] / IN_DIM;
    const int E = in_sizes[5];
    float* out = (float*)d_out;

    // Workspace layout (256B aligned slices)
    char* ws = (char*)d_ws;
    auto align_up = [](size_t v) { return (v + 255) & ~(size_t)255; };
    size_t o = 0;
    float* XW1  = (float*)(ws + o); o += align_up((size_t)N * HID * 4);
    float* Hraw = (float*)(ws + o); o += align_up((size_t)N * HID * 4);
    float* Z2   = (float*)(ws + o); o += align_up((size_t)N * 4);
    v16bf* bhi  = (v16bf*)(ws + o); o += align_up((size_t)1024 * 32);
    v16bf* blo  = (v16bf*)(ws + o); o += align_up((size_t)1024 * 32);
    (void)ws_size; (void)n_in; (void)out_size;

    // Independent setup work
    gcn_prep_w1_frags<<<4, 256, 0, stream>>>(W1, bhi, blo);
    gcn_zero_f32<<<2048, 256, 0, stream>>>(Hraw, (long long)N * HID);
    gcn_zero_f32<<<256, 256, 0, stream>>>(out, (long long)N);

    // Layer 1 dense: XW1 = X@W1 + b1 (bf16 hi/lo split WMMA, fp32 acc)
    gcn_gemm1_wmma<<<(N + 63) / 64, 128, 0, stream>>>(X, b1, bhi, blo, XW1, N);

    // Layer 1 sparse aggregate (L2-resident atomics)
    gcn_spmm1<<<8192, 256, 0, stream>>>(XW1, ew, esrc, edst, Hraw, E);

    // ReLU + layer 2 dense (wave reduction)
    gcn_relu_gemm2<<<(N + 7) / 8, 256, 0, stream>>>(Hraw, W2, b2, Z2, N);

    // Layer 2 sparse aggregate
    gcn_spmm2<<<8192, 256, 0, stream>>>(Z2, ew, esrc, edst, out, E);
}
